// NMT_45973329936755
// MI455X (gfx1250) — compile-verified, matrix operand order checked by reading, not run on hardware
//
#include <hip/hip_runtime.h>
#include <cstdint>
#include <cstddef>

// ---------------------------------------------------------------------------
// Types / helpers
// ---------------------------------------------------------------------------
typedef __attribute__((ext_vector_type(16))) __bf16 v16bf;
typedef __attribute__((ext_vector_type(8)))  __bf16 v8bf;
typedef __attribute__((ext_vector_type(8)))  float  v8f;

__device__ __forceinline__ __bf16 f2bf(float f) {
  unsigned u = __builtin_bit_cast(unsigned, f);
  u += 0x7FFFu + ((u >> 16) & 1u);              // round-to-nearest-even
  unsigned short s = (unsigned short)(u >> 16);
  return __builtin_bit_cast(__bf16, s);
}
__device__ __forceinline__ float bf2f(__bf16 b) {
  unsigned short s = __builtin_bit_cast(unsigned short, b);
  unsigned u = ((unsigned)s) << 16;
  return __builtin_bit_cast(float, u);
}

union FragBF { v16bf v; v8bf h[2]; };

// Load one WMMA operand fragment (bf16, K-major, documented layout:
// lane holds K = [hi*8 .. hi*8+7] and [16+hi*8 .. 16+hi*8+7]).
__device__ __forceinline__ v16bf ld_frag(const __bf16* p) {
  FragBF u;
  u.h[0] = *(const v8bf*)(p);
  u.h[1] = *(const v8bf*)(p + 16);
  return u.v;
}

#define WMMA_BF16(Af, Bf, Cacc) \
  __builtin_amdgcn_wmma_f32_16x16x32_bf16(false, (Af), false, (Bf), (short)0, (Cacc), false, false)

// ---------------------------------------------------------------------------
// Generic GEMM:  C(MxN, f32 and/or bf16) = A(MxK bf16, lda) @ B(NxK bf16)^T + bias
// REQUIRES: M % 32 == 0, N % 256 == 0, K % 32 == 0 (all call sites satisfy).
// Block: 256 threads = 8 waves; wave tile 16x64 (4 independent accumulators).
// Manual double buffering: iteration i loads fragment set i+1 while the WMMAs
// consume set i, so load->use distance is a full iteration (latency hidden).
// ---------------------------------------------------------------------------
__global__ __launch_bounds__(256)
void k_gemm_bf16(float* __restrict__ outF, __bf16* __restrict__ outBF,
                 const __bf16* __restrict__ A, int lda,
                 const __bf16* __restrict__ Bm,        // N x K row-major
                 const float* __restrict__ bias,
                 int K, int ldc)
{
  const int lane = threadIdx.x & 31;
  const int wave = __builtin_amdgcn_readfirstlane(threadIdx.x >> 5); // SGPR
  const int row  = lane & 15;
  const int hi   = lane >> 4;

  const int m0 = blockIdx.y * 32 + (wave >> 2) * 16;
  const int nb = blockIdx.x * 256 + (wave & 3) * 64;

  v8f acc0, acc1, acc2, acc3;
#pragma unroll
  for (int i = 0; i < 8; ++i) { acc0[i] = 0.f; acc1[i] = 0.f; acc2[i] = 0.f; acc3[i] = 0.f; }

  const __bf16* arow  = A + (size_t)(m0 + row) * lda + hi * 8;
  const __bf16* brow0 = Bm + (size_t)(nb + row) * K + hi * 8;
  const __bf16* brow1 = brow0 + (size_t)16 * K;
  const __bf16* brow2 = brow1 + (size_t)16 * K;
  const __bf16* brow3 = brow2 + (size_t)16 * K;

  // Prologue: fragment set for k0 = 0.
  v16bf af = ld_frag(arow);
  v16bf b0 = ld_frag(brow0);
  v16bf b1 = ld_frag(brow1);
  v16bf b2 = ld_frag(brow2);
  v16bf b3 = ld_frag(brow3);

#pragma unroll 2
  for (int k0 = 32; k0 < K; k0 += 32) {
    // Issue next set's loads before consuming the current set.
    v16bf afn = ld_frag(arow  + k0);
    v16bf b0n = ld_frag(brow0 + k0);
    v16bf b1n = ld_frag(brow1 + k0);
    v16bf b2n = ld_frag(brow2 + k0);
    v16bf b3n = ld_frag(brow3 + k0);
    __builtin_prefetch(arow + k0 + 32, 0, 1);   // speculative global_prefetch_b8
    acc0 = WMMA_BF16(af, b0, acc0);
    acc1 = WMMA_BF16(af, b1, acc1);
    acc2 = WMMA_BF16(af, b2, acc2);
    acc3 = WMMA_BF16(af, b3, acc3);
    af = afn; b0 = b0n; b1 = b1n; b2 = b2n; b3 = b3n;
  }
  // Tail: consume final set.
  acc0 = WMMA_BF16(af, b0, acc0);
  acc1 = WMMA_BF16(af, b1, acc1);
  acc2 = WMMA_BF16(af, b2, acc2);
  acc3 = WMMA_BF16(af, b3, acc3);

  // Epilogue: uniform scalar branches hoisted out of the unrolled loops.
  const int rbase = m0 + hi * 8;
  v8f* accs[4] = { &acc0, &acc1, &acc2, &acc3 };
  if (outF) {
#pragma unroll
    for (int s = 0; s < 4; ++s) {
      int col = nb + s * 16 + row;
      float bv = bias ? bias[col] : 0.0f;
#pragma unroll
      for (int j = 0; j < 8; ++j)
        outF[(size_t)(rbase + j) * ldc + col] = (*accs[s])[j] + bv;
    }
  }
  if (outBF) {
#pragma unroll
    for (int s = 0; s < 4; ++s) {
      int col = nb + s * 16 + row;
      float bv = bias ? bias[col] : 0.0f;
#pragma unroll
      for (int j = 0; j < 8; ++j)
        outBF[(size_t)(rbase + j) * ldc + col] = f2bf((*accs[s])[j] + bv);
    }
  }
}

// ---------------------------------------------------------------------------
// Elementwise / gather kernels
// ---------------------------------------------------------------------------
__global__ void k_cast_bf16(__bf16* o, const float* s, int n) {
  int i = blockIdx.x * blockDim.x + threadIdx.x;
  if (i < n) o[i] = f2bf(s[i]);
}

__global__ void k_cat2_bf16(__bf16* o, const float* a, int ka,
                            const float* b2, int kb, int n) {
  int i = blockIdx.x * blockDim.x + threadIdx.x;
  if (i >= n) return;
  int kk = ka + kb;
  int r = i / kk, k = i - r * kk;
  float v = (k < ka) ? a[(size_t)r * ka + k] : b2[(size_t)r * kb + (k - ka)];
  o[i] = f2bf(v);
}

__global__ void k_addvec(float* o, const float* a, const float* b, int n) {
  int i = blockIdx.x * blockDim.x + threadIdx.x;
  if (i < n) o[i] = a[i] + b[i];
}

__global__ void k_zero(float* p, int n) {
  int i = blockIdx.x * blockDim.x + threadIdx.x;
  if (i < n) p[i] = 0.0f;
}

__global__ void k_embed_bf16(__bf16* o, const int* toks, const float* table,
                             int rows, int E) {
  int i = blockIdx.x * blockDim.x + threadIdx.x;
  if (i >= rows * E) return;
  int r = i / E, e = i - r * E;
  o[i] = f2bf(table[(size_t)toks[r] * E + e]);
}

// LSTM cell: g = Xg (+ hWhh); gates [i|f|g|o] chunks of H; optional mask by t<len.
__global__ void k_lstm_cell(const float* __restrict__ Xg,
                            const float* __restrict__ hWhh,
                            float* __restrict__ h, float* __restrict__ c,
                            __bf16* __restrict__ h_bf, int hbf_stride,
                            __bf16* __restrict__ c_bf,
                            __bf16* __restrict__ out_bf, int out_bstride,
                            const int* __restrict__ lengths, int t)
{
  int i = blockIdx.x * blockDim.x + threadIdx.x;
  if (i >= 64 * 1024) return;
  int b = i >> 10, j = i & 1023;
  size_t gb = (size_t)b * 4096;
  float gi = Xg[gb + j],        gf = Xg[gb + 1024 + j];
  float gg = Xg[gb + 2048 + j], go = Xg[gb + 3072 + j];
  if (hWhh) {
    gi += hWhh[gb + j];        gf += hWhh[gb + 1024 + j];
    gg += hWhh[gb + 2048 + j]; go += hWhh[gb + 3072 + j];
  }
  float cp = c[i], hp = h[i];
  float si = 1.0f / (1.0f + expf(-gi));
  float sf = 1.0f / (1.0f + expf(-gf));
  float so = 1.0f / (1.0f + expf(-go));
  float c2 = sf * cp + si * tanhf(gg);
  float h2 = so * tanhf(c2);
  bool  m  = lengths ? (t < lengths[b]) : true;
  float hn = m ? h2 : hp;
  float cn = m ? c2 : cp;
  h[i] = hn; c[i] = cn;
  h_bf[(size_t)b * hbf_stride + j] = f2bf(hn);
  if (c_bf)  c_bf[(size_t)b * hbf_stride + j]  = f2bf(cn);
  if (out_bf) out_bf[(size_t)b * out_bstride + j] = f2bf(m ? h2 : 0.0f);
}

// Attention for one decoder step: scores, masked softmax over T=64, context a,
// and build abar = [a (2H) | h2 (H)] in bf16 for the W_comb GEMM.
__global__ __launch_bounds__(256)
void k_attention(__bf16* __restrict__ abar,             // B x 3072
                 const __bf16* __restrict__ enc_proj,   // (b*64+t)*1024
                 const __bf16* __restrict__ enc_h,      // (b*64+t)*2048
                 const float* __restrict__ h2,          // B x 1024
                 const int* __restrict__ lengths)
{
  int b = blockIdx.x, tid = threadIdx.x;
  __shared__ float part[256];
  __shared__ float alpha[64];
  int t = tid >> 2, q = tid & 3;
  const __bf16* ep = enc_proj + ((size_t)b * 64 + t) * 1024 + q * 256;
  const float*  hh = h2 + (size_t)b * 1024 + q * 256;
  float s = 0.0f;
  for (int j = 0; j < 256; ++j) s += bf2f(ep[j]) * hh[j];
  part[tid] = s;
  __syncthreads();
  if (q == 0) {
    float e = part[tid] + part[tid + 1] + part[tid + 2] + part[tid + 3];
    alpha[t] = (t < lengths[b]) ? e : -3.0e38f;
  }
  __syncthreads();
  if (tid == 0) {
    float mx = -3.0e38f;
    for (int i2 = 0; i2 < 64; ++i2) mx = fmaxf(mx, alpha[i2]);
    float sum = 0.0f;
    for (int i2 = 0; i2 < 64; ++i2) { float ex = expf(alpha[i2] - mx); alpha[i2] = ex; sum += ex; }
    part[0] = 1.0f / sum;
  }
  __syncthreads();
  float inv = part[0];
  for (int d = tid; d < 2048; d += 256) {
    const __bf16* eh = enc_h + (size_t)b * 64 * 2048 + d;
    float a = 0.0f;
    for (int i2 = 0; i2 < 64; ++i2) a += alpha[i2] * bf2f(eh[(size_t)i2 * 2048]);
    abar[(size_t)b * 3072 + d] = f2bf(a * inv);
  }
  for (int j = tid; j < 1024; j += 256)
    abar[(size_t)b * 3072 + 2048 + j] = f2bf(h2[(size_t)b * 1024 + j]);
}

__global__ void k_tanh_o(float* o_prev, __bf16* outs_row, const float* Og, int n) {
  int i = blockIdx.x * blockDim.x + threadIdx.x;
  if (i >= n) return;
  float v = tanhf(Og[i]);
  o_prev[i] = v;
  outs_row[i] = f2bf(v);
}

__global__ void k_zbar(__bf16* zbar, const __bf16* y, const float* o_prev,
                       const __bf16* h_bf, int n) {
  int i = blockIdx.x * blockDim.x + threadIdx.x;
  if (i >= n) return;
  int b = i / 2560, k = i - b * 2560;
  __bf16 v;
  if (k < 512)        v = y[(size_t)b * 512 + k];
  else if (k < 1536)  v = f2bf(o_prev[(size_t)b * 1024 + (k - 512)]);
  else                v = h_bf[(size_t)b * 1024 + (k - 1536)];
  zbar[i] = v;
}

// ---------------------------------------------------------------------------
// Fused vocab GEMM + online log-softmax + gold gather + masked time-sum.
// Block handles 16 rows of O (4032 x 1024); 8 waves stream 64-col groups of
// W_vocab (32000 x 1024) with WMMA, maintaining per-lane (max, sumexp).
// Never materializes the 4032x32000 logits (saves ~0.5 GB of HBM traffic).
// ---------------------------------------------------------------------------
__global__ __launch_bounds__(256)
void k_vocab_score(float* __restrict__ out,             // 64 accumulators
                   const __bf16* __restrict__ O,        // 4032 x 1024
                   const __bf16* __restrict__ Wv,       // 32000 x 1024
                   const int* __restrict__ target)      // 64*64 flat (T_TGT,B)
{
  const int Kd = 1024, Nv = 32000;
  const int lane = threadIdx.x & 31;
  const int wave = __builtin_amdgcn_readfirstlane(threadIdx.x >> 5); // SGPR
  const int row  = lane & 15;
  const int hi   = lane >> 4;
  const int rbase = blockIdx.x * 16;

  float m_run[8], s_run[8], g_val[8];
  int   g_tok[8];
#pragma unroll
  for (int j = 0; j < 8; ++j) {
    m_run[j] = -3.0e38f; s_run[j] = 0.0f; g_val[j] = -3.0e38f;
    g_tok[j] = target[rbase + hi * 8 + j + 64];   // target_padded[1:] flattened
  }

  const __bf16* arow = O + (size_t)(rbase + row) * Kd + hi * 8;
  const int NG = Nv / 64;                          // 500 groups of 64 columns
  for (int ng = wave; ng < NG; ng += 8) {          // scalar loop (wave in SGPR)
    int n0 = ng * 64;
    const __bf16* brow0 = Wv + (size_t)(n0 + row) * Kd + hi * 8;
    const __bf16* brow1 = brow0 + (size_t)16 * Kd;
    const __bf16* brow2 = brow1 + (size_t)16 * Kd;
    const __bf16* brow3 = brow2 + (size_t)16 * Kd;
    v8f acc0, acc1, acc2, acc3;
#pragma unroll
    for (int i2 = 0; i2 < 8; ++i2) { acc0[i2] = 0.f; acc1[i2] = 0.f; acc2[i2] = 0.f; acc3[i2] = 0.f; }

    v16bf af = ld_frag(arow);
    v16bf b0 = ld_frag(brow0);
    v16bf b1 = ld_frag(brow1);
    v16bf b2 = ld_frag(brow2);
    v16bf b3 = ld_frag(brow3);
#pragma unroll 2
    for (int k0 = 32; k0 < Kd; k0 += 32) {
      v16bf afn = ld_frag(arow  + k0);
      v16bf b0n = ld_frag(brow0 + k0);
      v16bf b1n = ld_frag(brow1 + k0);
      v16bf b2n = ld_frag(brow2 + k0);
      v16bf b3n = ld_frag(brow3 + k0);
      acc0 = WMMA_BF16(af, b0, acc0);
      acc1 = WMMA_BF16(af, b1, acc1);
      acc2 = WMMA_BF16(af, b2, acc2);
      acc3 = WMMA_BF16(af, b3, acc3);
      af = afn; b0 = b0n; b1 = b1n; b2 = b2n; b3 = b3n;
    }
    acc0 = WMMA_BF16(af, b0, acc0);
    acc1 = WMMA_BF16(af, b1, acc1);
    acc2 = WMMA_BF16(af, b2, acc2);
    acc3 = WMMA_BF16(af, b3, acc3);

    v8f* accs[4] = { &acc0, &acc1, &acc2, &acc3 };
#pragma unroll
    for (int s = 0; s < 4; ++s) {
      int col = n0 + s * 16 + row;
#pragma unroll
      for (int j = 0; j < 8; ++j) {
        float v  = (*accs[s])[j];
        float nm = fmaxf(m_run[j], v);
        s_run[j] = s_run[j] * expf(m_run[j] - nm) + expf(v - nm);
        m_run[j] = nm;
        if (col == g_tok[j]) g_val[j] = v;
      }
    }
  }

  // Merge the 256 partial (max, sumexp) streams per row via LDS.
  __shared__ float shm[16][257];
  __shared__ float shs[16][257];
  __shared__ float shg[16];
  int tid = threadIdx.x;
  for (int r16 = 0; r16 < 16; ++r16) { shm[r16][tid] = -3.0e38f; shs[r16][tid] = 0.0f; }
  if (tid < 16) shg[tid] = -3.0e38f;
  __syncthreads();
#pragma unroll
  for (int j = 0; j < 8; ++j) {
    shm[hi * 8 + j][tid] = m_run[j];
    shs[hi * 8 + j][tid] = s_run[j];
    if (g_val[j] > -1.0e37f) shg[hi * 8 + j] = g_val[j];   // unique writer per row
  }
  __syncthreads();
  if (tid < 16) {
    float M = -3.0e38f, S = 0.0f;
    for (int i2 = 0; i2 < 256; ++i2) {
      float m2 = shm[tid][i2], s2 = shs[tid][i2];
      float nm = fmaxf(M, m2);
      S = S * expf(M - nm) + s2 * expf(m2 - nm);
      M = nm;
    }
    int r = rbase + tid;
    float lp = shg[tid] - (M + logf(S));             // log_softmax at gold index
    int tok = target[r + 64];
    if (tok != 0) atomicAdd(&out[r & 63], lp);       // sum over time into batch b
  }
}

// ---------------------------------------------------------------------------
// Host orchestration
// ---------------------------------------------------------------------------
static inline dim3 gemm_grid(int M, int N) {
  return dim3((unsigned)(N / 256), (unsigned)(M / 32));
}
static inline int nb(int n) { return (n + 255) / 256; }

extern "C" void kernel_launch(void* const* d_in, const int* in_sizes, int n_in,
                              void* d_out, int out_size, void* d_ws, size_t ws_size,
                              hipStream_t stream) {
  (void)in_sizes; (void)n_in; (void)out_size; (void)ws_size;
  const int E = 512, H = 1024, G = 4096, Bz = 64, T = 64, TD = 63, V = 32000;
  const int R = TD * Bz;                    // 4032 decoder rows

  const int*   src_tok = (const int*)d_in[0];
  const int*   tgt_tok = (const int*)d_in[1];
  const int*   lens    = (const int*)d_in[2];
  const float* src_emb = (const float*)d_in[3];
  const float* dst_emb = (const float*)d_in[4];
  const float* Wihf = (const float*)d_in[5];
  const float* Whhf = (const float*)d_in[6];
  const float* bihf = (const float*)d_in[7];
  const float* bhhf = (const float*)d_in[8];
  const float* Wihb = (const float*)d_in[9];
  const float* Whhb = (const float*)d_in[10];
  const float* bihb = (const float*)d_in[11];
  const float* bhhb = (const float*)d_in[12];
  const float* Wihd = (const float*)d_in[13];
  const float* Whhd = (const float*)d_in[14];
  const float* bihd = (const float*)d_in[15];
  const float* bhhd = (const float*)d_in[16];
  const float* Whp  = (const float*)d_in[17];
  const float* Wcp  = (const float*)d_in[18];
  const float* Watt = (const float*)d_in[19];
  const float* Wcmb = (const float*)d_in[20];
  const float* Wvoc = (const float*)d_in[21];
  float* out = (float*)d_out;

  char* ws = (char*)d_ws;
  size_t off = 0;
  auto alloc = [&](size_t elems, size_t esz) -> char* {
    char* p = ws + off;
    off = (off + elems * esz + 255) & ~(size_t)255;
    return p;
  };

  __bf16* wihf_bf  = (__bf16*)alloc((size_t)G * E, 2);
  __bf16* whhf_bf  = (__bf16*)alloc((size_t)G * H, 2);
  __bf16* wihb_bf  = (__bf16*)alloc((size_t)G * E, 2);
  __bf16* whhb_bf  = (__bf16*)alloc((size_t)G * H, 2);
  __bf16* wdcat_bf = (__bf16*)alloc((size_t)G * 2560, 2);
  __bf16* whp_bf   = (__bf16*)alloc((size_t)H * 2 * H, 2);
  __bf16* wcp_bf   = (__bf16*)alloc((size_t)H * 2 * H, 2);
  __bf16* watt_bf  = (__bf16*)alloc((size_t)H * 2 * H, 2);
  __bf16* wcmb_bf  = (__bf16*)alloc((size_t)H * 3 * H, 2);
  __bf16* wvoc_bf  = (__bf16*)alloc((size_t)V * H, 2);
  float*  biasf    = (float*)alloc(G, 4);
  float*  biasb    = (float*)alloc(G, 4);
  float*  biasd    = (float*)alloc(G, 4);
  __bf16* xsrc_bf  = (__bf16*)alloc((size_t)T * Bz * E, 2);
  __bf16* yemb_bf  = (__bf16*)alloc((size_t)TD * Bz * E, 2);
  float*  xgf      = (float*)alloc((size_t)T * Bz * G, 4);
  float*  xgb      = (float*)alloc((size_t)T * Bz * G, 4);
  float*  g2       = (float*)alloc((size_t)Bz * G, 4);
  float*  hF = (float*)alloc((size_t)Bz * H, 4);
  float*  cF = (float*)alloc((size_t)Bz * H, 4);
  float*  hB = (float*)alloc((size_t)Bz * H, 4);
  float*  cB = (float*)alloc((size_t)Bz * H, 4);
  float*  hD = (float*)alloc((size_t)Bz * H, 4);
  float*  cD = (float*)alloc((size_t)Bz * H, 4);
  float*  oprev = (float*)alloc((size_t)Bz * H, 4);
  __bf16* hcat_bf = (__bf16*)alloc((size_t)Bz * 2 * H, 2);
  __bf16* ccat_bf = (__bf16*)alloc((size_t)Bz * 2 * H, 2);
  __bf16* hd_bf   = (__bf16*)alloc((size_t)Bz * H, 2);
  __bf16* ench_bf = (__bf16*)alloc((size_t)Bz * T * 2 * H, 2);
  __bf16* encp_bf = (__bf16*)alloc((size_t)Bz * T * H, 2);
  __bf16* zbar_bf = (__bf16*)alloc((size_t)Bz * 2560, 2);
  __bf16* abar_bf = (__bf16*)alloc((size_t)Bz * 3 * H, 2);
  float*  og      = (float*)alloc((size_t)Bz * H, 4);
  __bf16* outs_bf = (__bf16*)alloc((size_t)R * H, 2);

  // --- weight conversions (bf16) and bias sums ------------------------------
  k_cast_bf16<<<nb(G * E), 256, 0, stream>>>(wihf_bf, Wihf, G * E);
  k_cast_bf16<<<nb(G * H), 256, 0, stream>>>(whhf_bf, Whhf, G * H);
  k_cast_bf16<<<nb(G * E), 256, 0, stream>>>(wihb_bf, Wihb, G * E);
  k_cast_bf16<<<nb(G * H), 256, 0, stream>>>(whhb_bf, Whhb, G * H);
  k_cat2_bf16<<<nb(G * 2560), 256, 0, stream>>>(wdcat_bf, Wihd, E + H, Whhd, H, G * 2560);
  k_cast_bf16<<<nb(H * 2 * H), 256, 0, stream>>>(whp_bf, Whp, H * 2 * H);
  k_cast_bf16<<<nb(H * 2 * H), 256, 0, stream>>>(wcp_bf, Wcp, H * 2 * H);
  k_cast_bf16<<<nb(H * 2 * H), 256, 0, stream>>>(watt_bf, Watt, H * 2 * H);
  k_cast_bf16<<<nb(H * 3 * H), 256, 0, stream>>>(wcmb_bf, Wcmb, H * 3 * H);
  k_cast_bf16<<<nb(V * H), 256, 0, stream>>>(wvoc_bf, Wvoc, V * H);
  k_addvec<<<nb(G), 256, 0, stream>>>(biasf, bihf, bhhf, G);
  k_addvec<<<nb(G), 256, 0, stream>>>(biasb, bihb, bhhb, G);
  k_addvec<<<nb(G), 256, 0, stream>>>(biasd, bihd, bhhd, G);

  // --- embeddings (gather + cast in one pass) -------------------------------
  k_embed_bf16<<<nb(T * Bz * E), 256, 0, stream>>>(xsrc_bf, src_tok, src_emb, T * Bz, E);
  k_embed_bf16<<<nb(TD * Bz * E), 256, 0, stream>>>(yemb_bf, tgt_tok, dst_emb, TD * Bz, E);

  // --- bulk input-side GEMMs: Xg = X @ W_ih^T + (b_ih + b_hh) ---------------
  k_gemm_bf16<<<gemm_grid(T * Bz, G), 256, 0, stream>>>(
      xgf, nullptr, xsrc_bf, E, wihf_bf, biasf, E, G);
  k_gemm_bf16<<<gemm_grid(T * Bz, G), 256, 0, stream>>>(
      xgb, nullptr, xsrc_bf, E, wihb_bf, biasb, E, G);

  // --- state init -----------------------------------------------------------
  k_zero<<<nb(Bz * H), 256, 0, stream>>>(hF, Bz * H);
  k_zero<<<nb(Bz * H), 256, 0, stream>>>(cF, Bz * H);
  k_zero<<<nb(Bz * H), 256, 0, stream>>>(hB, Bz * H);
  k_zero<<<nb(Bz * H), 256, 0, stream>>>(cB, Bz * H);
  k_zero<<<nb(Bz * H), 256, 0, stream>>>((float*)hcat_bf, Bz * H);  // bf16 zeros
  k_zero<<<nb(Bz * H), 256, 0, stream>>>((float*)ccat_bf, Bz * H);

  // --- encoder forward scan -------------------------------------------------
  for (int t = 0; t < T; ++t) {
    k_gemm_bf16<<<gemm_grid(Bz, G), 256, 0, stream>>>(
        g2, nullptr, hcat_bf, 2 * H, whhf_bf, nullptr, H, G);
    k_lstm_cell<<<nb(Bz * H), 256, 0, stream>>>(
        xgf + (size_t)t * Bz * G, g2, hF, cF,
        hcat_bf, 2 * H, ccat_bf,
        ench_bf + (size_t)t * 2 * H, T * 2 * H, lens, t);
  }
  // --- encoder backward scan ------------------------------------------------
  for (int s = 0; s < T; ++s) {
    int t = T - 1 - s;
    k_gemm_bf16<<<gemm_grid(Bz, G), 256, 0, stream>>>(
        g2, nullptr, hcat_bf + H, 2 * H, whhb_bf, nullptr, H, G);
    k_lstm_cell<<<nb(Bz * H), 256, 0, stream>>>(
        xgb + (size_t)t * Bz * G, g2, hB, cB,
        hcat_bf + H, 2 * H, ccat_bf + H,
        ench_bf + (size_t)t * 2 * H + H, T * 2 * H, lens, t);
  }

  // --- decoder init: h0 = [hf,hb]@Whp^T ; c0 = [cf,cb]@Wcp^T ----------------
  k_gemm_bf16<<<gemm_grid(Bz, H), 256, 0, stream>>>(
      hD, hd_bf, hcat_bf, 2 * H, whp_bf, nullptr, 2 * H, H);
  k_gemm_bf16<<<gemm_grid(Bz, H), 256, 0, stream>>>(
      cD, nullptr, ccat_bf, 2 * H, wcp_bf, nullptr, 2 * H, H);

  // --- enc_proj = enc_hiddens @ W_att^T (bf16 out) --------------------------
  k_gemm_bf16<<<gemm_grid(Bz * T, H), 256, 0, stream>>>(
      nullptr, encp_bf, ench_bf, 2 * H, watt_bf, nullptr, 2 * H, H);

  k_zero<<<nb(Bz * H), 256, 0, stream>>>(oprev, Bz * H);

  // --- decoder scan ---------------------------------------------------------
  for (int t = 0; t < TD; ++t) {
    k_zbar<<<nb(Bz * 2560), 256, 0, stream>>>(
        zbar_bf, yemb_bf + (size_t)t * Bz * E, oprev, hd_bf, Bz * 2560);
    k_gemm_bf16<<<gemm_grid(Bz, G), 256, 0, stream>>>(
        g2, nullptr, zbar_bf, 2560, wdcat_bf, biasd, 2560, G);
    k_lstm_cell<<<nb(Bz * H), 256, 0, stream>>>(
        g2, nullptr, hD, cD, hd_bf, H, nullptr, nullptr, 0, nullptr, 0);
    k_attention<<<Bz, 256, 0, stream>>>(abar_bf, encp_bf, ench_bf, hD, lens);
    k_gemm_bf16<<<gemm_grid(Bz, H), 256, 0, stream>>>(
        og, nullptr, abar_bf, 3 * H, wcmb_bf, nullptr, 3 * H, H);
    k_tanh_o<<<nb(Bz * H), 256, 0, stream>>>(
        oprev, outs_bf + (size_t)t * Bz * H, og, Bz * H);
  }

  // --- fused vocab projection + log-softmax + gold gather + time-sum --------
  k_zero<<<1, 64, 0, stream>>>(out, Bz);
  k_vocab_score<<<R / 16, 256, 0, stream>>>(out, outs_bf, wvoc_bf, tgt_tok);
}
